// Llama4TextMoELayer_87222195847675
// MI455X (gfx1250) — matrix operand order repaired and checked by reading, not compiled, not run on hardware
//
#include <hip/hip_runtime.h>
#include <cstdint>
#include <cstddef>

// ---------------- problem constants ----------------
constexpr int kSeq = 2048, kBs = 4, kTok = kSeq * kBs;   // 8192 tokens
constexpr int kH = 2048;      // hidden
constexpr int kI = 4096;      // intermediate
constexpr int kE = 8;         // experts
constexpr int kPermCap = 8320; // >= kTok + kE*15, multiple of 16
constexpr int kBStride = 40;  // LDS row stride (bf16) for transposed B slices: 80B, 16B-aligned

// ---------------- types ----------------
typedef __bf16 bf16_t;
typedef bf16_t v4bf  __attribute__((ext_vector_type(4)));
typedef bf16_t v8bf  __attribute__((ext_vector_type(8)));
typedef bf16_t v16bf __attribute__((ext_vector_type(16)));
typedef float  v4f   __attribute__((ext_vector_type(4)));
typedef float  v8f   __attribute__((ext_vector_type(8)));

__device__ __forceinline__ v4bf cvt4(v4f x) {
  return __builtin_convertvector(x, v4bf);     // packed v_cvt_pk_bf16_f32
}
__device__ __forceinline__ v16bf cat8(v8bf lo, v8bf hi) {
  return __builtin_shufflevector(lo, hi, 0,1,2,3,4,5,6,7,8,9,10,11,12,13,14,15);
}

// Stage one [32 x 128] fp32 weight slice (row stride ld) transposed into LDS as
// bf16 [n][k] with padded row stride kBStride. 128 threads.
__device__ __forceinline__ void stage_w(const float* __restrict__ W, size_t ld,
                                        bf16_t* __restrict__ lB,
                                        int kk, int n0, int tid) {
#pragma unroll
  for (int p = 0; p < 8; ++p) {
    int v  = p * 128 + tid;
    int kr = v >> 5;            // 0..31 (k row)
    int nc = (v & 31) * 4;      // 0..124 (n quad)
    v4f w = *(const v4f*)(W + (size_t)(kk + kr) * ld + n0 + nc);
    v4bf b = cvt4(w);
    lB[(nc + 0) * kBStride + kr] = b[0];
    lB[(nc + 1) * kBStride + kr] = b[1];
    lB[(nc + 2) * kBStride + kr] = b[2];
    lB[(nc + 3) * kBStride + kr] = b[3];
  }
}

// ---------------- kernel 1: init workspace ----------------
__global__ void moe_init_kernel(int* cnt, int* cnt2, int* perm) {
  int t = blockIdx.x * blockDim.x + threadIdx.x;
  if (t < kPermCap) perm[t] = -1;
  if (t < kE) { cnt[t] = 0; cnt2[t] = 0; }
}

// ---------------- kernel 2: router (wave per token) ----------------
__global__ void moe_router_kernel(const float* __restrict__ hs,
                                  const float* __restrict__ rw,
                                  int* __restrict__ idx, int* __restrict__ cnt) {
  const int lane = threadIdx.x & 31;
  const int wv   = threadIdx.x >> 5;
  const int t    = blockIdx.x * 8 + wv;            // 1024 blocks * 8 waves
  if (t >= kTok) return;

  float acc[kE] = {0.f,0.f,0.f,0.f,0.f,0.f,0.f,0.f};
  const float* hrow = hs + (size_t)t * kH;
  for (int h = lane; h < kH; h += 32) {
    float x = hrow[h];
    const float4* wp = (const float4*)(rw + (size_t)h * kE);   // E=8 -> 2x float4
    float4 w0 = wp[0], w1 = wp[1];
    acc[0] += x * w0.x; acc[1] += x * w0.y; acc[2] += x * w0.z; acc[3] += x * w0.w;
    acc[4] += x * w1.x; acc[5] += x * w1.y; acc[6] += x * w1.z; acc[7] += x * w1.w;
  }
#pragma unroll
  for (int s = 16; s > 0; s >>= 1) {
#pragma unroll
    for (int e = 0; e < kE; ++e) acc[e] += __shfl_xor(acc[e], s, 32);
  }
  if (lane == 0) {
    int best = 0; float bv = acc[0];
#pragma unroll
    for (int e = 1; e < kE; ++e) if (acc[e] > bv) { bv = acc[e]; best = e; }
    idx[t] = best;
    atomicAdd(&cnt[best], 1);
  }
}

// ---------------- kernel 3: 16-aligned prefix offsets ----------------
__global__ void moe_offsets_kernel(const int* __restrict__ cnt,
                                   int* __restrict__ offs, int* __restrict__ tiles) {
  if (threadIdx.x == 0 && blockIdx.x == 0) {
    int o = 0;
    for (int e = 0; e < kE; ++e) {
      int nt = (cnt[e] + 15) >> 4;
      tiles[e] = nt;
      offs[e]  = o;
      o += nt * 16;
    }
    offs[kE] = o;
  }
}

// ---------------- kernel 4: scatter tokens into expert-sorted order ----------------
__global__ void moe_scatter_kernel(const int* __restrict__ idx,
                                   const int* __restrict__ offs,
                                   int* __restrict__ cnt2, int* __restrict__ perm) {
  int t = blockIdx.x * blockDim.x + threadIdx.x;
  if (t >= kTok) return;
  int e = idx[t];
  int slot = atomicAdd(&cnt2[e], 1);
  perm[offs[e] + slot] = t;
}

// ---------------- kernel 5: gate+up GEMM + SiLU (bf16 WMMA, double-buffered) ----------------
// grid: (512 tiles, kI/128 nchunks, kE experts), block: 128 threads = 4 waves
// Each wave owns 32 N-cols (two 16x16 WMMA subtiles) per matrix.
__global__ void __launch_bounds__(128)
moe_gateup_kernel(const float* __restrict__ hs,
                  const float* __restrict__ gate_w,
                  const float* __restrict__ up_w,
                  const int* __restrict__ perm,
                  const int* __restrict__ offs,
                  const int* __restrict__ tiles,
                  bf16_t* __restrict__ act) {
  const int e     = blockIdx.z;
  const int tile  = blockIdx.x;
  if (tile >= tiles[e]) return;
  const int n0      = blockIdx.y * 128;
  const int rowbase = offs[e] + tile * 16;

  __shared__ int rowTok[16];
  __shared__ bf16_t lA[2][16 * 32];           // [m][k]
  __shared__ bf16_t lG[2][128 * kBStride];    // [n][k] transposed, padded
  __shared__ bf16_t lU[2][128 * kBStride];

  const int tid  = threadIdx.x;
  const int lane = tid & 31;
  const int wv   = tid >> 5;        // 0..3
  const int hi   = lane >> 4;       // half-wave
  const int m    = lane & 15;

  if (tid < 16) rowTok[tid] = perm[rowbase + tid];
  __syncthreads();

  // A-staging assignment (all 128 threads): one float4 per K-step
  const int  amr  = tid >> 3;
  const int  aj4  = (tid & 7) * 4;
  const int  atok = rowTok[amr];
  const float* aptr = (atok >= 0) ? (hs + (size_t)atok * kH + aj4) : nullptr;

  const float* GW = gate_w + (size_t)e * kH * kI;
  const float* UW = up_w   + (size_t)e * kH * kI;

  v8f cg0 = {}, cg1 = {}, cu0 = {}, cu1 = {};

  // prologue: stage k0 = 0 into buffer 0
  {
    v4f x = {};
    if (aptr) x = *(const v4f*)(aptr);
    *(v4bf*)&lA[0][amr * 32 + aj4] = cvt4(x);
    stage_w(GW, kI, lG[0], 0, n0, tid);
    stage_w(UW, kI, lU[0], 0, n0, tid);
  }
  __syncthreads();

  int cur = 0;
  for (int k0 = 0; k0 < kH; k0 += 32) {
    // stage next slice into the other buffer (overlaps with WMMA below)
    if (k0 + 32 < kH) {
      const int nxt = cur ^ 1, kk = k0 + 32;
      v4f x = {};
      if (aptr) x = *(const v4f*)(aptr + kk);
      *(v4bf*)&lA[nxt][amr * 32 + aj4] = cvt4(x);
      stage_w(GW, kI, lG[nxt], kk, n0, tid);
      stage_w(UW, kI, lU[nxt], kk, n0, tid);
      if (kk + 32 < kH) {
        __builtin_prefetch(GW + (size_t)(kk + 32) * kI + n0, 0, 1);
        __builtin_prefetch(UW + (size_t)(kk + 32) * kI + n0, 0, 1);
      }
    }

    // consume current buffer: A and B fragments in contiguous k order (k = 16*hi + s)
    v8bf a0 = *(const v8bf*)&lA[cur][m * 32 + hi * 16];
    v8bf a1 = *(const v8bf*)&lA[cur][m * 32 + hi * 16 + 8];
    v16bf A = cat8(a0, a1);

    const int nb0 = (wv * 32 + m) * kBStride + hi * 16;
    const int nb1 = (wv * 32 + 16 + m) * kBStride + hi * 16;

    v16bf G0 = cat8(*(const v8bf*)&lG[cur][nb0], *(const v8bf*)&lG[cur][nb0 + 8]);
    v16bf G1 = cat8(*(const v8bf*)&lG[cur][nb1], *(const v8bf*)&lG[cur][nb1 + 8]);
    v16bf U0 = cat8(*(const v8bf*)&lU[cur][nb0], *(const v8bf*)&lU[cur][nb0 + 8]);
    v16bf U1 = cat8(*(const v8bf*)&lU[cur][nb1], *(const v8bf*)&lU[cur][nb1 + 8]);

    cg0 = __builtin_amdgcn_wmma_f32_16x16x32_bf16(false, A, false, G0, (short)0, cg0, false, false);
    cg1 = __builtin_amdgcn_wmma_f32_16x16x32_bf16(false, A, false, G1, (short)0, cg1, false, false);
    cu0 = __builtin_amdgcn_wmma_f32_16x16x32_bf16(false, A, false, U0, (short)0, cu0, false, false);
    cu1 = __builtin_amdgcn_wmma_f32_16x16x32_bf16(false, A, false, U1, (short)0, cu1, false, false);

    __syncthreads();
    cur ^= 1;
  }

  // --- SiLU(g)*u, write bf16 activation rows (sorted-slot indexed) ---
  const int col0 = n0 + wv * 32 + m;
#pragma unroll
  for (int r = 0; r < 8; ++r) {
    int mm = r + hi * 8;
    size_t rowoff = (size_t)(rowbase + mm) * kI;
    float g0 = cg0[r], u0 = cu0[r];
    float g1 = cg1[r], u1 = cu1[r];
    float a0 = (g0 / (1.f + __expf(-g0))) * u0;
    float a1 = (g1 / (1.f + __expf(-g1))) * u1;
    act[rowoff + col0]      = (bf16_t)a0;
    act[rowoff + col0 + 16] = (bf16_t)a1;
  }
}

// ---------------- kernel 6: down GEMM + scatter to out (double-buffered) ----------------
// grid: (512 tiles, kH/128 hchunks, kE experts), block: 128 threads = 4 waves
__global__ void __launch_bounds__(128)
moe_down_kernel(const bf16_t* __restrict__ act,
                const float* __restrict__ down_w,
                const int* __restrict__ perm,
                const int* __restrict__ offs,
                const int* __restrict__ tiles,
                float* __restrict__ out) {
  const int e    = blockIdx.z;
  const int tile = blockIdx.x;
  if (tile >= tiles[e]) return;
  const int h0      = blockIdx.y * 128;
  const int rowbase = offs[e] + tile * 16;

  __shared__ int rowTok[16];
  __shared__ bf16_t lD[2][128 * kBStride];    // [n(h)][k] transposed, padded

  const int tid  = threadIdx.x;
  const int lane = tid & 31;
  const int wv   = tid >> 5;
  const int hi   = lane >> 4;
  const int m    = lane & 15;

  if (tid < 16) rowTok[tid] = perm[rowbase + tid];
  __syncthreads();

  const float* DW = down_w + (size_t)e * kI * kH;
  const bf16_t* Arow = act + (size_t)(rowbase + m) * kI;   // this lane's A row

  v8f c0 = {}, c1 = {};

  stage_w(DW, kH, lD[0], 0, h0, tid);
  __syncthreads();

  int cur = 0;
  for (int k0 = 0; k0 < kI; k0 += 32) {
    if (k0 + 32 < kI) {
      stage_w(DW, kH, lD[cur ^ 1], k0 + 32, h0, tid);
      if (k0 + 64 < kI)
        __builtin_prefetch(DW + (size_t)(k0 + 64) * kH + h0, 0, 1);
    }

    // A fragment straight from bf16 act rows (global_load_b128 x2)
    v8bf a0 = *(const v8bf*)(Arow + k0 + hi * 16);
    v8bf a1 = *(const v8bf*)(Arow + k0 + hi * 16 + 8);
    v16bf A = cat8(a0, a1);

    const int nb0 = (wv * 32 + m) * kBStride + hi * 16;
    const int nb1 = (wv * 32 + 16 + m) * kBStride + hi * 16;
    v16bf B0 = cat8(*(const v8bf*)&lD[cur][nb0], *(const v8bf*)&lD[cur][nb0 + 8]);
    v16bf B1 = cat8(*(const v8bf*)&lD[cur][nb1], *(const v8bf*)&lD[cur][nb1 + 8]);

    c0 = __builtin_amdgcn_wmma_f32_16x16x32_bf16(false, A, false, B0, (short)0, c0, false, false);
    c1 = __builtin_amdgcn_wmma_f32_16x16x32_bf16(false, A, false, B1, (short)0, c1, false, false);

    __syncthreads();
    cur ^= 1;
  }

  // --- scatter rows to out (each token owned by exactly one expert slot) ---
  const int col0 = h0 + wv * 32 + m;
#pragma unroll
  for (int r = 0; r < 8; ++r) {
    int mm = r + hi * 8;
    int t = rowTok[mm];
    if (t >= 0) {
      out[(size_t)t * kH + col0]      = c0[r];
      out[(size_t)t * kH + col0 + 16] = c1[r];
    }
  }
}

// ---------------- host launcher ----------------
extern "C" void kernel_launch(void* const* d_in, const int* in_sizes, int n_in,
                              void* d_out, int out_size, void* d_ws, size_t ws_size,
                              hipStream_t stream) {
  const float* hs     = (const float*)d_in[0];   // [S,B,H]
  const float* rw     = (const float*)d_in[1];   // [H,E]
  const float* gate_w = (const float*)d_in[2];   // [E,H,I]
  const float* up_w   = (const float*)d_in[3];   // [E,H,I]
  const float* down_w = (const float*)d_in[4];   // [E,I,H]
  float* out          = (float*)d_out;           // [S,B,H]

  // workspace layout (256B aligned regions)
  char* ws = (char*)d_ws;
  int* idx   = (int*)(ws + 0);                         // 8192 ints
  int* cnt   = (int*)(ws + 32768);                     // 8
  int* cnt2  = (int*)(ws + 33024);                     // 8
  int* offs  = (int*)(ws + 33280);                     // 9
  int* tiles = (int*)(ws + 33536);                     // 8
  int* perm  = (int*)(ws + 33792);                     // kPermCap ints
  bf16_t* act = (bf16_t*)(ws + 67584);                 // kPermCap * kI bf16 (~68 MB)

  moe_init_kernel<<<dim3((kPermCap + 255) / 256), dim3(256), 0, stream>>>(cnt, cnt2, perm);
  moe_router_kernel<<<dim3(kTok / 8), dim3(256), 0, stream>>>(hs, rw, idx, cnt);
  moe_offsets_kernel<<<dim3(1), dim3(1), 0, stream>>>(cnt, offs, tiles);
  moe_scatter_kernel<<<dim3(kTok / 256), dim3(256), 0, stream>>>(idx, offs, cnt2, perm);
  moe_gateup_kernel<<<dim3(kTok / 16, kI / 128, kE), dim3(128), 0, stream>>>(
      hs, gate_w, up_w, perm, offs, tiles, act);
  moe_down_kernel<<<dim3(kTok / 16, kH / 128, kE), dim3(128), 0, stream>>>(
      act, down_w, perm, offs, tiles, out);
}
